// RelScaleAttend_82652350644354
// MI455X (gfx1250) — matrix operand order repaired and checked by reading, not compiled, not run on hardware
//
#include <hip/hip_runtime.h>
#include <hip/hip_bf16.h>

typedef __attribute__((ext_vector_type(16))) _Float16 v16h;
typedef __attribute__((ext_vector_type(8)))  float    v8f;

#define S_LEN 1024
#define HEADS_TOTAL 64   // b(4) * n_heads(16)
#define DHEAD 64
#define LOG2E 1.44269504088896f
#define NJ 16            // 16 key blocks of 64 keys

__global__ __launch_bounds__(128) void
fa_relpos_kernel(const float* __restrict__ q, const float* __restrict__ k,
                 const float* __restrict__ v, const float* __restrict__ rph,
                 const float* __restrict__ rpw, float* __restrict__ out)
{
    // Double-buffered async staging (raw f32, filled by global_load_async_to_lds)
    __shared__ __align__(32) float    Kst[2][64][68];  // [key][d] f32
    __shared__ __align__(32) float    Vst[2][64][68];  // [key][d] f32
    __shared__ __align__(32) _Float16 Kh[64][80];      // [key][d] f16 (shared cvt)
    __shared__ __align__(32) _Float16 Vt[64][80];      // [d][key] f16 (transposed)
    __shared__ __align__(32) float    Qs[4][16][68];   // per-wave fp32 Q tile
    __shared__ __align__(32) float    Bh[4][16][32];   // rel-h bias (log2-domain)
    __shared__ __align__(32) _Float16 Pl[4][16][80];   // per-wave P staging

    const int tid  = threadIdx.x;
    const int lane = tid & 31;
    const int wv   = tid >> 5;
    const int l16  = lane & 15;
    const int hi   = lane >> 4;

    const int head = blockIdx.x >> 4;
    const int qt   = blockIdx.x & 15;
    const int b    = head >> 4;
    const int h    = head & 15;
    const size_t base = (size_t)b * S_LEN * 1024 + (size_t)h * DHEAD;
    const int wqbase = qt * 64 + wv * 16;

    const float* kbase = k + base;
    const float* vbase = v + base;

    // ---- async K/V block prefetch: 64 keys x 64 d, 16B per lane-issue ----
    auto issue_block = [&](int jj, int bufsel) {
        unsigned kst0 = (unsigned)(uintptr_t)&Kst[bufsel][0][0];
        unsigned vst0 = (unsigned)(uintptr_t)&Vst[bufsel][0][0];
        #pragma unroll
        for (int i = 0; i < 8; ++i) {
            int lin = tid + 128 * i;            // 0..1023 b128 chunks
            int key = lin >> 4, dc = lin & 15;
            unsigned voff = (unsigned)((jj * 64 + key) * 4096 + dc * 16);
            unsigned loff = (unsigned)(key * 272 + dc * 16);
            asm volatile("global_load_async_to_lds_b128 %0, %1, %2"
                         :: "v"(kst0 + loff), "v"(voff),
                            "s"((unsigned long long)(uintptr_t)kbase) : "memory");
            asm volatile("global_load_async_to_lds_b128 %0, %1, %2"
                         :: "v"(vst0 + loff), "v"(voff),
                            "s"((unsigned long long)(uintptr_t)vbase) : "memory");
        }
    };

    issue_block(0, 0);   // overlap first transfer with the whole prologue

    // ---- stage Q tile (fp32) ----
    for (int idx = lane; idx < 16 * DHEAD; idx += 32) {
        int r = idx >> 6, dd = idx & 63;
        Qs[wv][r][dd] = q[base + (size_t)(wqbase + r) * 1024 + dd];
    }
    __syncthreads();

    // ---- rel-pos biases, pre-scaled by log2(e) ----
    v8f bw0{}, bw1{};
    #pragma unroll
    for (int r = 0; r < 8; ++r) {
        const int M    = r + 8 * hi;
        const int qrow = wqbase + M;
        const int hq   = (qrow >> 5) & 31;
        const int wq   = qrow & 31;
        #pragma unroll
        for (int n = 0; n < 2; ++n) {
            const int kc = n * 16 + l16;
            const float* tw = rpw + (size_t)(wq - kc + 31) * DHEAD;
            const float* th = rph + (size_t)(hq - kc + 31) * DHEAD;
            float accw = 0.f, acch = 0.f;
            #pragma unroll 4
            for (int c = 0; c < DHEAD; ++c) {
                float qc = Qs[wv][M][c];
                accw += qc * tw[c];
                acch += qc * th[c];
            }
            if (n == 0) bw0[r] = accw * LOG2E; else bw1[r] = accw * LOG2E;
            Bh[wv][M][kc] = acch * LOG2E;
        }
    }

    // ---- Q A-fragments (16x32 f16 x2) ----
    v16h aq0, aq1;
    {
        const int d0 = 16 * hi;
        v16h a0, a1;
        #pragma unroll
        for (int i = 0; i < 8; ++i) {
            a0[i]     = (_Float16)Qs[wv][l16][d0 + i];
            a0[8 + i] = (_Float16)Qs[wv][l16][d0 + 8 + i];
            a1[i]     = (_Float16)Qs[wv][l16][32 + d0 + i];
            a1[8 + i] = (_Float16)Qs[wv][l16][32 + d0 + 8 + i];
        }
        aq0 = a0; aq1 = a1;
    }

    // ones-column B fragment: col 0 = 1.0 -> row sums of P via WMMA
    v16h bones;
    #pragma unroll
    for (int i = 0; i < 16; ++i) bones[i] = (_Float16)((l16 == 0) ? 1.0f : 0.0f);

    float m8[8];
    v8f oarr[4], oext{};
    #pragma unroll
    for (int n = 0; n < 4; ++n) oarr[n] = v8f{};
    #pragma unroll
    for (int r = 0; r < 8; ++r) m8[r] = -1e30f;

    const float scl2 = 0.125f * LOG2E;   // 64^-0.5 in log2 domain

    for (int j = 0; j < NJ; ++j) {
        const int buf = j & 1;
        asm volatile("s_wait_asynccnt 0x0" ::: "memory");  // my transfers for buf done
        __syncthreads();                                   // all done; prev consumers done
        if (j + 1 < NJ) issue_block(j + 1, buf ^ 1);       // overlap next block

        // ---- shared cvt pass: Kst f32 -> Kh f16 (same layout) ----
        {
            int key = tid >> 1, dh = (tid & 1) * 32;
            const float* src = &Kst[buf][key][dh];
            v16h h0, h1;
            #pragma unroll
            for (int i = 0; i < 16; ++i) { h0[i] = (_Float16)src[i]; h1[i] = (_Float16)src[16 + i]; }
            *(v16h*)&Kh[key][dh]      = h0;
            *(v16h*)&Kh[key][dh + 16] = h1;
        }
        // ---- transpose pass: Vst f32 [key][d] -> Vt f16 [d][key] ----
        #pragma unroll
        for (int g = 0; g < 2; ++g) {
            int lin = tid * 2 + g;          // 0..255 : (d, 16-key group)
            int dd = lin >> 2, kg = lin & 3;
            v16h hv;
            #pragma unroll
            for (int kk = 0; kk < 16; ++kk)
                hv[kk] = (_Float16)Vst[buf][kg * 16 + kk][dd];
            *(v16h*)&Vt[dd][kg * 16] = hv;
        }
        __syncthreads();

        // ---- S = Q K^T : 4 col-fragments x K=64 -> 8 WMMAs ----
        v8f sarr[4];
        #pragma unroll
        for (int n = 0; n < 4; ++n) {
            v16h b0 = *(const v16h*)&Kh[n * 16 + l16][16 * hi];
            v16h b1 = *(const v16h*)&Kh[n * 16 + l16][32 + 16 * hi];
            v8f acc{};
            acc = __builtin_amdgcn_wmma_f32_16x16x32_f16(false, aq0, false, b0, (short)0, acc, false, false);
            acc = __builtin_amdgcn_wmma_f32_16x16x32_f16(false, aq1, false, b1, (short)0, acc, false, false);
            sarr[n] = acc;
        }

        // ---- bias + online softmax (log2 domain, max-only reduction) ----
        float corr[8];
        #pragma unroll
        for (int r = 0; r < 8; ++r) {
            const int M = r + 8 * hi;
            float bh0 = Bh[wv][M][2 * j];
            float bh1 = Bh[wv][M][2 * j + 1];
            float a0 = sarr[0][r] * scl2 + bh0 + bw0[r];
            float a1 = sarr[1][r] * scl2 + bh0 + bw1[r];
            float a2 = sarr[2][r] * scl2 + bh1 + bw0[r];
            float a3 = sarr[3][r] * scl2 + bh1 + bw1[r];
            float mx = fmaxf(fmaxf(a0, a1), fmaxf(a2, a3));
            #pragma unroll
            for (int off = 1; off < 16; off <<= 1)
                mx = fmaxf(mx, __shfl_xor(mx, off, 32));
            float nm = fmaxf(m8[r], mx);
            float cr = __builtin_amdgcn_exp2f(m8[r] - nm);
            m8[r] = nm;
            corr[r] = cr;
            Pl[wv][M][l16]      = (_Float16)__builtin_amdgcn_exp2f(a0 - nm);
            Pl[wv][M][16 + l16] = (_Float16)__builtin_amdgcn_exp2f(a1 - nm);
            Pl[wv][M][32 + l16] = (_Float16)__builtin_amdgcn_exp2f(a2 - nm);
            Pl[wv][M][48 + l16] = (_Float16)__builtin_amdgcn_exp2f(a3 - nm);
        }
        #pragma unroll
        for (int r = 0; r < 8; ++r) {
            oarr[0][r] *= corr[r]; oarr[1][r] *= corr[r];
            oarr[2][r] *= corr[r]; oarr[3][r] *= corr[r];
            oext[r]    *= corr[r];
        }

        // ---- P -> A layout (K=64: two 16x32 chunks) ----
        asm volatile("s_wait_dscnt 0" ::: "memory");
        v16h ap0, ap1;
        #pragma unroll
        for (int i = 0; i < 8; ++i) {
            ap0[i]     = Pl[wv][l16][16 * hi + i];
            ap0[8 + i] = Pl[wv][l16][8 + 16 * hi + i];
            ap1[i]     = Pl[wv][l16][32 + 16 * hi + i];
            ap1[8 + i] = Pl[wv][l16][40 + 16 * hi + i];
        }

        // ---- O += P V (8 WMMAs) ; row sums += P @ ones (2 WMMAs) ----
        #pragma unroll
        for (int n = 0; n < 4; ++n) {
            v16h bv0 = *(const v16h*)&Vt[n * 16 + l16][16 * hi];
            v16h bv1 = *(const v16h*)&Vt[n * 16 + l16][32 + 16 * hi];
            oarr[n] = __builtin_amdgcn_wmma_f32_16x16x32_f16(false, ap0, false, bv0, (short)0, oarr[n], false, false);
            oarr[n] = __builtin_amdgcn_wmma_f32_16x16x32_f16(false, ap1, false, bv1, (short)0, oarr[n], false, false);
        }
        oext = __builtin_amdgcn_wmma_f32_16x16x32_f16(false, ap0, false, bones, (short)0, oext, false, false);
        oext = __builtin_amdgcn_wmma_f32_16x16x32_f16(false, ap1, false, bones, (short)0, oext, false, false);
    }

    // ---- normalize and store (denominator lives in lane l16==0 of each half) ----
    #pragma unroll
    for (int r = 0; r < 8; ++r) {
        const int M = r + 8 * hi;
        float denom = __shfl(oext[r], hi * 16, 32);
        float inv = 1.0f / denom;
        size_t rowp = base + (size_t)(wqbase + M) * 1024;
        out[rowp + 0  + l16] = oarr[0][r] * inv;
        out[rowp + 16 + l16] = oarr[1][r] * inv;
        out[rowp + 32 + l16] = oarr[2][r] * inv;
        out[rowp + 48 + l16] = oarr[3][r] * inv;
    }
}

extern "C" void kernel_launch(void* const* d_in, const int* in_sizes, int n_in,
                              void* d_out, int out_size, void* d_ws, size_t ws_size,
                              hipStream_t stream) {
    const float* q   = (const float*)d_in[0];
    const float* k   = (const float*)d_in[1];
    const float* v   = (const float*)d_in[2];
    const float* rph = (const float*)d_in[3];
    const float* rpw = (const float*)d_in[4];
    float* out = (float*)d_out;

    dim3 grid(HEADS_TOTAL * (S_LEN / 64));  // 1024 workgroups
    dim3 block(128);                        // 4 waves (wave32)
    hipLaunchKernelGGL(fa_relpos_kernel, grid, block, 0, stream,
                       q, k, v, rph, rpw, out);
}
